// PointTransformerLayer_82377472737949
// MI455X (gfx1250) — matrix operand order repaired and checked by reading, not compiled, not run on hardware
//
#include <hip/hip_runtime.h>

// Point Transformer layer for MI455X (gfx1250, wave32, WMMA f32<-f16 16x16x32).
// B=4, N=4096, d_points=256, d_model=64, K=16 (fixed by the reference).

typedef __attribute__((ext_vector_type(16))) _Float16 v16h;
typedef __attribute__((ext_vector_type(8)))  _Float16 v8h;
typedef __attribute__((ext_vector_type(8)))  float    v8f;

#define NPTS  4096
#define BATCH 4
#define NTOT  (NPTS * BATCH)   // 16384 points total
#define DM    64               // d_model
#define DP    256              // d_points
#define KNBR  16               // neighbors

// ---- WMMA fragment index maps (ISA 7.12.2, wave32) -------------------------
// 16-bit A (16x32): lane<16 -> row=lane%16, K = {0..7,16..23}; lane>=16 same
// row, K = {8..15,24..31}.  => K-contiguous in two groups of 8.
// 16-bit B (32x16): lane%16 = column; same K split as A: lane<16 K={0..7,16..23}, lane>=16 K={8..15,24..31}.

// A fragment from row-major fp32 global memory: 4 x b128 loads + cvt.
__device__ __forceinline__ v16h a_frag_g(const float* __restrict__ A, int lda,
                                         int row0, int kb, int lane) {
  const float* r = A + (size_t)(row0 + (lane & 15)) * lda + kb + ((lane & 16) ? 8 : 0);
  const float4 p0 = *(const float4*)(r + 0);
  const float4 p1 = *(const float4*)(r + 4);
  const float4 p2 = *(const float4*)(r + 16);
  const float4 p3 = *(const float4*)(r + 20);
  v16h f;
  f[0]  = (_Float16)p0.x; f[1]  = (_Float16)p0.y; f[2]  = (_Float16)p0.z; f[3]  = (_Float16)p0.w;
  f[4]  = (_Float16)p1.x; f[5]  = (_Float16)p1.y; f[6]  = (_Float16)p1.z; f[7]  = (_Float16)p1.w;
  f[8]  = (_Float16)p2.x; f[9]  = (_Float16)p2.y; f[10] = (_Float16)p2.z; f[11] = (_Float16)p2.w;
  f[12] = (_Float16)p3.x; f[13] = (_Float16)p3.y; f[14] = (_Float16)p3.z; f[15] = (_Float16)p3.w;
  return f;
}

// B fragment from K-contiguous (transposed) f16 weights in LDS: WT[N][Kld].
// Two groups of 8 halves per lane (K = 8h..8h+7, 16+8h..23+8h) -> two ds b128 loads.
__device__ __forceinline__ v16h b_frag_t(const _Float16* WT, int Kld, int kb,
                                         int col0, int lane) {
  const _Float16* p = WT + (size_t)(col0 + (lane & 15)) * Kld + kb + ((lane & 16) ? 8 : 0);
  const v8h lo = *(const v8h*)p;
  const v8h hi = *(const v8h*)(p + 16);
  return __builtin_shufflevector(lo, hi, 0, 1, 2, 3, 4, 5, 6, 7, 8, 9, 10, 11, 12, 13, 14, 15);
}

// A fragment from a 16 x ld f16 tile in LDS (row-major): two ds b128 loads.
__device__ __forceinline__ v16h a_frag_lds_h(const _Float16* T, int ld, int kb, int lane) {
  const _Float16* r = T + (lane & 15) * ld + kb + ((lane & 16) ? 8 : 0);
  const v8h lo = *(const v8h*)r;
  const v8h hi = *(const v8h*)(r + 16);
  return __builtin_shufflevector(lo, hi, 0, 1, 2, 3, 4, 5, 6, 7, 8, 9, 10, 11, 12, 13, 14, 15);
}

// A fragment from a 16 x ld f32 tile in LDS (two contiguous groups of 8).
__device__ __forceinline__ v16h a_frag_lds_f(const float* T, int ld, int kb, int lane) {
  const float* r = T + (lane & 15) * ld + kb + ((lane & 16) ? 8 : 0);
  v16h f;
#pragma unroll
  for (int h = 0; h < 8; ++h) { f[h] = (_Float16)r[h]; f[h + 8] = (_Float16)r[16 + h]; }
  return f;
}

// Cooperative transpose-stage: W[K][NCOLS] fp32 (coalesced read) -> WT[NCOLS][K] f16.
template <int NCOLS>
__device__ __forceinline__ void stage_wt(const float* __restrict__ W, _Float16* WT,
                                         int K, int tid) {
  const int total = K * NCOLS;
  for (int i = tid; i < total; i += 256) {
    const int k = i / NCOLS, n = i % NCOLS;
    WT[n * K + k] = (_Float16)W[i];
  }
}

#define WMMA(A, B, C) \
  __builtin_amdgcn_wmma_f32_16x16x32_f16(false, (A), false, (B), (short)0, (C), false, false)

// ============================================================================
// Kernel 1: brute-force KNN. xyz of one batch (48 KB) lives in LDS; each
// thread owns one query and keeps a sorted top-16 in registers.
// ============================================================================
__global__ __launch_bounds__(256) void knn_kernel(const float* __restrict__ xyz,
                                                  int* __restrict__ knn_idx) {
  __shared__ float sx[NPTS], sy[NPTS], sz[NPTS];
  const int b = blockIdx.y;
  const float* xb = xyz + (size_t)b * NPTS * 3;
  for (int j = threadIdx.x; j < NPTS; j += 256) {
    sx[j] = xb[j * 3 + 0];
    sy[j] = xb[j * 3 + 1];
    sz[j] = xb[j * 3 + 2];
  }
  __syncthreads();

  const int i = blockIdx.x * 256 + threadIdx.x;
  const float qx = sx[i], qy = sy[i], qz = sz[i];
  float bd[KNBR];
  int   bi[KNBR];
#pragma unroll
  for (int t = 0; t < KNBR; ++t) { bd[t] = 3.0e38f; bi[t] = 0; }

  for (int j = 0; j < NPTS; ++j) {
    const float dx = qx - sx[j], dy = qy - sy[j], dz = qz - sz[j];
    const float d = dx * dx + dy * dy + dz * dz;
    if (d < bd[KNBR - 1]) {             // rare -> cheap amortized insert
      bd[KNBR - 1] = d; bi[KNBR - 1] = j;
#pragma unroll
      for (int t = KNBR - 1; t > 0; --t) {
        if (bd[t] < bd[t - 1]) {
          float td = bd[t]; bd[t] = bd[t - 1]; bd[t - 1] = td;
          int   ti = bi[t]; bi[t] = bi[t - 1]; bi[t - 1] = ti;
        }
      }
    }
  }
  int* o = knn_idx + ((size_t)b * NPTS + i) * KNBR;
#pragma unroll
  for (int t = 0; t < KNBR; ++t) o[t] = bi[t];
}

// ============================================================================
// Kernel 2: x = fc1(features); then q,k,v = x @ {wq,wk,wv}.
// Weights staged transposed f16 in (dynamic) LDS once per block; x tile staged
// f16 per wave. One wave -> 16 rows.
// Dynamic LDS layout (halves): fc1t[64*256] | wqt|wkt|wvt[64*64] | xtile[8*16*64]
// ============================================================================
__global__ __launch_bounds__(256) void qkv_kernel(
    const float* __restrict__ feats,
    const float* __restrict__ fc1w, const float* __restrict__ fc1b,
    const float* __restrict__ wq, const float* __restrict__ wk, const float* __restrict__ wv,
    float* __restrict__ q, float* __restrict__ kk, float* __restrict__ vv) {
  extern __shared__ _Float16 smem[];
  _Float16* fc1t  = smem;                     // 64 x 256
  _Float16* wqt   = fc1t + DM * DP;           // 64 x 64
  _Float16* wkt   = wqt + DM * DM;
  _Float16* wvt   = wkt + DM * DM;
  _Float16* xtile = wvt + DM * DM;            // 8 waves x (16 x 64)

  const int tid = threadIdx.x;
  stage_wt<DM>(fc1w, fc1t, DP, tid);
  stage_wt<DM>(wq, wqt, DM, tid);
  stage_wt<DM>(wk, wkt, DM, tid);
  stage_wt<DM>(wv, wvt, DM, tid);
  __syncthreads();

  const int wave = tid >> 5;
  const int lane = tid & 31;
  const int hi = lane >> 4, cl = lane & 15;
  const int row0 = (blockIdx.x * 8 + wave) * 16;
  _Float16* xt = xtile + wave * 16 * DM;

  // x = features @ fc1_w + fc1_b  (M=16, K=256, N=64)
  v8f acc[4];
#pragma unroll
  for (int t = 0; t < 4; ++t) {
    const float bv = fc1b[t * 16 + cl];
#pragma unroll
    for (int r = 0; r < 8; ++r) acc[t][r] = bv;
  }
  for (int kc = 0; kc < DP; kc += 32) {
    const v16h a = a_frag_g(feats, DP, row0, kc, lane);
#pragma unroll
    for (int t = 0; t < 4; ++t) {
      const v16h bf = b_frag_t(fc1t, DP, kc, t * 16, lane);
      acc[t] = WMMA(a, bf, acc[t]);
    }
  }
#pragma unroll
  for (int t = 0; t < 4; ++t)
#pragma unroll
    for (int r = 0; r < 8; ++r)
      xt[(r + 8 * hi) * DM + t * 16 + cl] = (_Float16)acc[t][r];

  // q/k/v projections (M=16, K=64, N=64 each); intra-wave LDS ops are in-order.
  const _Float16* Wt[3] = {wqt, wkt, wvt};
  float*          Os[3] = {q, kk, vv};
#pragma unroll
  for (int w = 0; w < 3; ++w) {
    v8f c[4];
#pragma unroll
    for (int t = 0; t < 4; ++t)
#pragma unroll
      for (int r = 0; r < 8; ++r) c[t][r] = 0.0f;
    for (int kc = 0; kc < DM; kc += 32) {
      const v16h a = a_frag_lds_h(xt, DM, kc, lane);
#pragma unroll
      for (int t = 0; t < 4; ++t) {
        const v16h bf = b_frag_t(Wt[w], DM, kc, t * 16, lane);
        c[t] = WMMA(a, bf, c[t]);
      }
    }
#pragma unroll
    for (int t = 0; t < 4; ++t)
#pragma unroll
      for (int r = 0; r < 8; ++r)
        Os[w][(size_t)(row0 + r + 8 * hi) * DM + t * 16 + cl] = c[t][r];
  }
}

// ============================================================================
// Kernel 3: fused per-query attention, one wave per point:
//   pos  = relu(rel@d1+b1)@d2+b2              (VALU 3->64, WMMA 64->64)
//   attn = relu((q-k+pos)@g1+b1)@g2+b2        (2x WMMA 64->64)
//   attn = softmax over K neighbors; L1-normalize == identity post-softmax
//   res  = sum_k attn * (v + pos)
// d2/g1/g2 staged transposed f16 in LDS once per block; D-layout -> A-layout
// transposes go through a per-wave f32 LDS tile.
// ============================================================================
__global__ __launch_bounds__(256) void attn_kernel(
    const float* __restrict__ xyz, const int* __restrict__ knn_idx,
    const float* __restrict__ q, const float* __restrict__ kk, const float* __restrict__ vv,
    const float* __restrict__ d1w, const float* __restrict__ d1b,
    const float* __restrict__ d2w, const float* __restrict__ d2b,
    const float* __restrict__ g1w, const float* __restrict__ g1b,
    const float* __restrict__ g2w, const float* __restrict__ g2b,
    float* __restrict__ res) {
  __shared__ float    tile[8][16 * DM];      // 32 KB
  __shared__ float    sq[8][DM];             // 2 KB
  __shared__ float    srel[8][KNBR * 3];
  __shared__ int      sidx[8][KNBR];
  __shared__ _Float16 d2t[DM * DM];          // 8 KB each, transposed f16
  __shared__ _Float16 g1t[DM * DM];
  __shared__ _Float16 g2t[DM * DM];
  __shared__ float    sd1[3 * DM];           // d1_w
  __shared__ float    sb[4 * DM];            // d1_b | d2_b | g1_b | g2_b

  const int tid = threadIdx.x;
  stage_wt<DM>(d2w, d2t, DM, tid);
  stage_wt<DM>(g1w, g1t, DM, tid);
  stage_wt<DM>(g2w, g2t, DM, tid);
  if (tid < 3 * DM) sd1[tid] = d1w[tid];
  if (tid < DM) {
    sb[tid]          = d1b[tid];
    sb[DM + tid]     = d2b[tid];
    sb[2 * DM + tid] = g1b[tid];
    sb[3 * DM + tid] = g2b[tid];
  }
  __syncthreads();

  const int wave = tid >> 5;
  const int lane = tid & 31;
  const int hi = lane >> 4, cl = lane & 15;
  const int n = blockIdx.x * 8 + wave;      // global point id
  const int bofs = (n >> 12) << 12;         // batch row offset (N=4096)
  const int shift = (lane & 16) ? 8 : 0;    // A-fragment K shift for this lane

  if (lane < KNBR) {
    const int j = knn_idx[(size_t)n * KNBR + lane];
    sidx[wave][lane] = j;
    const float* pc = xyz + (size_t)n * 3;
    const float* pj = xyz + (size_t)(bofs + j) * 3;
    srel[wave][lane * 3 + 0] = pc[0] - pj[0];
    srel[wave][lane * 3 + 1] = pc[1] - pj[1];
    srel[wave][lane * 3 + 2] = pc[2] - pj[2];
  }
  sq[wave][lane]      = q[(size_t)n * DM + lane];
  sq[wave][lane + 32] = q[(size_t)n * DM + lane + 32];

  // pos-enc layer 1 (3 -> 64 + relu), built directly in A-fragment layout.
  const int m = cl;                          // this lane's neighbor row
  const float rx = srel[wave][m * 3 + 0];
  const float ry = srel[wave][m * 3 + 1];
  const float rz = srel[wave][m * 3 + 2];
  v16h Ap[2];
#pragma unroll
  for (int c = 0; c < 2; ++c) {
    const int base = c * 32 + shift;
#pragma unroll
    for (int h = 0; h < 16; ++h) {
      const int ch = base + ((h < 8) ? h : h + 8);
      const float v = rx * sd1[ch] + ry * sd1[DM + ch] + rz * sd1[2 * DM + ch] + sb[ch];
      Ap[c][h] = (_Float16)fmaxf(v, 0.0f);
    }
  }

  // pos-enc layer 2: 16x64 @ 64x64 + d2_b
  v8f pos[4];
#pragma unroll
  for (int t = 0; t < 4; ++t) {
    const float bv = sb[DM + t * 16 + cl];
#pragma unroll
    for (int r = 0; r < 8; ++r) pos[t][r] = bv;
  }
#pragma unroll
  for (int c = 0; c < 2; ++c)
#pragma unroll
    for (int t = 0; t < 4; ++t) {
      const v16h bf = b_frag_t(d2t, DM, c * 32, t * 16, lane);
      pos[t] = WMMA(Ap[c], bf, pos[t]);
    }

  // D-layout -> row-major staging for the next A operand.
#pragma unroll
  for (int t = 0; t < 4; ++t)
#pragma unroll
    for (int r = 0; r < 8; ++r)
      tile[wave][(r + 8 * hi) * DM + t * 16 + cl] = pos[t][r];

  // A = q - k(neighbor) + pos_enc   (k row gathered as 4x b128 per chunk)
  const int gi = bofs + sidx[wave][m];
  const float* krow = kk + (size_t)gi * DM;
  v16h Aa[2];
#pragma unroll
  for (int c = 0; c < 2; ++c) {
    const int base = c * 32 + shift;
    const float4 k0 = *(const float4*)(krow + base + 0);
    const float4 k1 = *(const float4*)(krow + base + 4);
    const float4 k2 = *(const float4*)(krow + base + 16);
    const float4 k3 = *(const float4*)(krow + base + 20);
    const float kv[16] = {k0.x, k0.y, k0.z, k0.w, k1.x, k1.y, k1.z, k1.w,
                          k2.x, k2.y, k2.z, k2.w, k3.x, k3.y, k3.z, k3.w};
#pragma unroll
    for (int h = 0; h < 16; ++h) {
      const int ch = base + ((h < 8) ? h : h + 8);
      Aa[c][h] = (_Float16)(sq[wave][ch] - kv[h] + tile[wave][m * DM + ch]);
    }
  }

  // gamma layer 1 (+ relu), restaged to LDS
  v8f hidv[4];
#pragma unroll
  for (int t = 0; t < 4; ++t) {
    const float bv = sb[2 * DM + t * 16 + cl];
#pragma unroll
    for (int r = 0; r < 8; ++r) hidv[t][r] = bv;
  }
#pragma unroll
  for (int c = 0; c < 2; ++c)
#pragma unroll
    for (int t = 0; t < 4; ++t) {
      const v16h bf = b_frag_t(g1t, DM, c * 32, t * 16, lane);
      hidv[t] = WMMA(Aa[c], bf, hidv[t]);
    }
#pragma unroll
  for (int t = 0; t < 4; ++t)
#pragma unroll
    for (int r = 0; r < 8; ++r)
      tile[wave][(r + 8 * hi) * DM + t * 16 + cl] = fmaxf(hidv[t][r], 0.0f);

  // gamma layer 2
  v8f att[4];
#pragma unroll
  for (int t = 0; t < 4; ++t) {
    const float bv = sb[3 * DM + t * 16 + cl];
#pragma unroll
    for (int r = 0; r < 8; ++r) att[t][r] = bv;
  }
#pragma unroll
  for (int c = 0; c < 2; ++c) {
    const v16h a = a_frag_lds_f(tile[wave], DM, c * 32, lane);
#pragma unroll
    for (int t = 0; t < 4; ++t) {
      const v16h bf = b_frag_t(g2t, DM, c * 32, t * 16, lane);
      att[t] = WMMA(a, bf, att[t]);
    }
  }

  // softmax over the 16 neighbors (rows of each 16x16 D tile): reduce over the
  // 8 row-VGPRs, then across the two lane-halves (lane ^ 16).
#pragma unroll
  for (int t = 0; t < 4; ++t) {
    float mx = att[t][0];
#pragma unroll
    for (int r = 1; r < 8; ++r) mx = fmaxf(mx, att[t][r]);
    mx = fmaxf(mx, __shfl_xor(mx, 16, 32));
    float s = 0.0f;
#pragma unroll
    for (int r = 0; r < 8; ++r) { const float e = __expf(att[t][r] - mx); att[t][r] = e; s += e; }
    s += __shfl_xor(s, 16, 32);
    const float inv = 1.0f / fmaxf(s, 1e-12f);   // L1 normalize == identity post-softmax
#pragma unroll
    for (int r = 0; r < 8; ++r) att[t][r] *= inv;
  }

  // res[ch] = sum_k attn[k,ch] * (v[k,ch] + pos[k,ch])
  int gr[8];
#pragma unroll
  for (int r = 0; r < 8; ++r) gr[r] = bofs + sidx[wave][r + 8 * hi];
#pragma unroll
  for (int t = 0; t < 4; ++t) {
    float acc = 0.0f;
#pragma unroll
    for (int r = 0; r < 8; ++r)
      acc += att[t][r] * (vv[(size_t)gr[r] * DM + t * 16 + cl] + pos[t][r]);
    acc += __shfl_xor(acc, 16, 32);
    if (lane < 16) res[(size_t)n * DM + t * 16 + lane] = acc;
  }
}

// ============================================================================
// Kernel 4: out = res @ fc2_w + fc2_b + features  (M=16384, K=64, N=256).
// fc2_w staged transposed f16 in LDS; one wave per 16x16 output tile.
// ============================================================================
__global__ __launch_bounds__(256) void out_kernel(
    const float* __restrict__ res, const float* __restrict__ feats,
    const float* __restrict__ fc2w, const float* __restrict__ fc2b,
    float* __restrict__ out) {
  __shared__ _Float16 fc2t[DP * DM];   // transposed: [256][64], 32 KB
  const int tid = threadIdx.x;
  stage_wt<DP>(fc2w, fc2t, DM, tid);
  __syncthreads();

  const int wave = tid >> 5;
  const int lane = tid & 31;
  const int hi = lane >> 4, cl = lane & 15;
  const int tile_id = blockIdx.x * 8 + wave;
  const int row0 = (tile_id >> 4) * 16;      // 16 col-tiles per row-block
  const int col0 = (tile_id & 15) * 16;

  v8f c;
  const float bv = fc2b[col0 + cl];
#pragma unroll
  for (int r = 0; r < 8; ++r) c[r] = bv;
  for (int kc = 0; kc < DM; kc += 32) {
    const v16h a  = a_frag_g(res, DM, row0, kc, lane);
    const v16h bf = b_frag_t(fc2t, DM, kc, col0, lane);
    c = WMMA(a, bf, c);
  }
#pragma unroll
  for (int r = 0; r < 8; ++r) {
    const size_t o = (size_t)(row0 + r + 8 * hi) * DP + col0 + cl;
    out[o] = c[r] + feats[o];
  }
}

// ============================================================================
extern "C" void kernel_launch(void* const* d_in, const int* in_sizes, int n_in,
                              void* d_out, int out_size, void* d_ws, size_t ws_size,
                              hipStream_t stream) {
  const float* xyz   = (const float*)d_in[0];
  const float* feats = (const float*)d_in[1];
  const float* fc1w  = (const float*)d_in[2];
  const float* fc1b  = (const float*)d_in[3];
  const float* fc2w  = (const float*)d_in[4];
  const float* fc2b  = (const float*)d_in[5];
  const float* d1w   = (const float*)d_in[6];
  const float* d1b   = (const float*)d_in[7];
  const float* d2w   = (const float*)d_in[8];
  const float* d2b   = (const float*)d_in[9];
  const float* g1w   = (const float*)d_in[10];
  const float* g1b   = (const float*)d_in[11];
  const float* g2w   = (const float*)d_in[12];
  const float* g2b   = (const float*)d_in[13];
  const float* wq    = (const float*)d_in[14];
  const float* wk    = (const float*)d_in[15];
  const float* wv    = (const float*)d_in[16];
  float* out = (float*)d_out;

  // workspace layout (17 MB): knn idx | q | k | v | res
  char* ws = (char*)d_ws;
  int*   knn = (int*)ws;        ws += (size_t)NTOT * KNBR * sizeof(int);
  float* q   = (float*)ws;      ws += (size_t)NTOT * DM * sizeof(float);
  float* kk  = (float*)ws;      ws += (size_t)NTOT * DM * sizeof(float);
  float* vv  = (float*)ws;      ws += (size_t)NTOT * DM * sizeof(float);
  float* res = (float*)ws;

  // dynamic LDS for qkv: fc1t + wq/wk/wv (transposed f16) + 8 x-tiles
  const size_t qkv_lds = (size_t)(DM * DP + 3 * DM * DM + 8 * 16 * DM) * sizeof(_Float16);

  knn_kernel <<<dim3(NPTS / 256, BATCH), 256, 0, stream>>>(xyz, knn);
  qkv_kernel <<<NTOT / 16 / 8, 256, qkv_lds, stream>>>(feats, fc1w, fc1b, wq, wk, wv, q, kk, vv);
  attn_kernel<<<NTOT / 8, 256, 0, stream>>>(xyz, knn, q, kk, vv,
                                            d1w, d1b, d2w, d2b, g1w, g1b, g2w, g2b, res);
  out_kernel <<<(NTOT / 16) * (DP / 16) / 8, 256, 0, stream>>>(res, feats, fc2w, fc2b, out);
}
